// VariationalAutoEncoder_48146583388406
// MI455X (gfx1250) — matrix operand order repaired and checked by reading, not compiled, not run on hardware
//
#include <hip/hip_runtime.h>
#include <math.h>

#define Bc 128
#define Tc 256
#define Uc 1024
#define Lc 256
#define Vc 512
#define NGATE 4096 /* 4*Uc */

typedef __attribute__((ext_vector_type(16))) __bf16 v16bf;
typedef __attribute__((ext_vector_type(8)))  float  v8f;

union FragBF { uint4 u[2]; v16bf v; };

__device__ __forceinline__ unsigned short f2bf(float x) {
  unsigned int u = __float_as_uint(x);
  unsigned int r = ((u >> 16) & 1u) + 0x7FFFu;     // round-to-nearest-even
  return (unsigned short)((u + r) >> 16);
}
__device__ __forceinline__ float sigf(float x) { return 1.0f / (1.0f + expf(-x)); }

// ---------------------------------------------------------------------------
// Fused LSTM step: one kernel does z = concat(A0|A1|A2) @ W^T + b for all four
// gates AND the pointwise gate/state update.
//
// Mapping: each wave owns a 16(M) x 16(u) patch but accumulates FOUR output
// tiles -- the same u-range in each gate block (n = g*1024 + u). One A
// fragment feeds 4 WMMAs (2 A-loads : 8 B-loads : 4 wmma per K-chunk), and at
// the end of the K-loop the wave holds i,f,g,o in registers, so c/h are
// updated in the epilogue with no z round-trip through memory.
// W is pre-transposed bf16: Bt[n][Ktot].
// grid = (128/16, 1024/64), block = 128 threads (4 waves).
// ---------------------------------------------------------------------------
__global__ void lstm_step_wmma(
    const unsigned short* __restrict__ A0, long lda0, int K0,
    const unsigned short* __restrict__ A1, long lda1, int K1,
    const unsigned short* __restrict__ A2, long lda2, int K2,
    const unsigned short* __restrict__ Bt, long ldb,
    const float* __restrict__ bias,
    float* __restrict__ c, unsigned short* __restrict__ h,
    unsigned short* __restrict__ y, long yRowStride)
{
  const int lane = threadIdx.x & 31;
  const int wave = threadIdx.x >> 5;
  const int m0   = blockIdx.x << 4;
  const int u0   = (blockIdx.y << 6) + (wave << 4);
  const int l15  = lane & 15;
  const int hi   = lane >> 4;

  const int row   = m0 + l15;
  const int halfA = hi << 3;           // A: lanes 16-31 hold K+8..
  const int halfB = hi << 3;           // B: lanes 16-31 hold K+8..

  v8f acc[4] = {{}, {}, {}, {}};
  const unsigned short* brow[4];
#pragma unroll
  for (int g = 0; g < 4; ++g)
    brow[g] = Bt + (long)(g * Uc + u0 + l15) * ldb + halfB;

  const unsigned short* segp[3] = {A0, A1, A2};
  const long segl[3] = {lda0, lda1, lda2};
  const int  segk[3] = {K0, K1, K2};

  int kglob = 0;
#pragma unroll
  for (int s = 0; s < 3; ++s) {
    const int Ks = segk[s];
    if (Ks <= 0) continue;
    const unsigned short* arow = segp[s] + (long)row * segl[s] + halfA;
    for (int k = 0; k < Ks; k += 32) {
      FragBF fa;
      fa.u[0] = *(const uint4*)(arow + k);
      fa.u[1] = *(const uint4*)(arow + k + 16);
#pragma unroll
      for (int g = 0; g < 4; ++g) {
        FragBF fb;
        const unsigned short* bp = brow[g] + kglob + k;
        fb.u[0] = *(const uint4*)(bp);
        fb.u[1] = *(const uint4*)(bp + 16);
        acc[g] = __builtin_amdgcn_wmma_f32_16x16x32_bf16(
            false, fa.v, false, fb.v, (short)0, acc[g], false, false);
      }
    }
    kglob += Ks;
  }

  // ---- fused gate epilogue -----------------------------------------------
  // C/D layout: VGPR r -> row m0 + r + (hi?8:0), col u0 + (lane&15)
  const int u  = u0 + l15;
  const int mb = m0 + (hi << 3);
  const float bi_ = bias[u];
  const float bf_ = bias[Uc + u];
  const float bg_ = bias[2 * Uc + u];
  const float bo_ = bias[3 * Uc + u];
#pragma unroll
  for (int r = 0; r < 8; ++r) {
    const int m = mb + r;
    const long ci = (long)m * Uc + u;
    float ig = sigf(acc[0][r] + bi_);
    float fg = sigf(acc[1][r] + bf_);
    float gg = tanhf(acc[2][r] + bg_);
    float og = sigf(acc[3][r] + bo_);
    float cn = fg * c[ci] + ig * gg;
    c[ci] = cn;
    unsigned short hb = f2bf(og * tanhf(cn));
    h[ci] = hb;
    if (y) y[(long)m * yRowStride + u] = hb;
  }
}

// ---------------------------------------------------------------------------
// Plain WMMA GEMM (single A segment) for the output projection:
// C[M,N](f32) = A[M,K]bf16 @ Bt[N,K]bf16^T + bias.
// grid = (M/16, N/64), block = 128 (4 waves, one 16x16 tile each).
// ---------------------------------------------------------------------------
__global__ void wmma_gemm_bf16(
    const unsigned short* __restrict__ A, long lda, int K,
    const unsigned short* __restrict__ Bt, long ldb,
    const float* __restrict__ bias,
    float* __restrict__ C, int N)
{
  const int lane = threadIdx.x & 31;
  const int wave = threadIdx.x >> 5;
  const int m0   = blockIdx.x << 4;
  const int n0   = (blockIdx.y << 6) + (wave << 4);
  const int l15  = lane & 15;
  const int hi   = lane >> 4;

  const unsigned short* arow = A + (long)(m0 + l15) * lda + (hi << 3);
  const unsigned short* brow = Bt + (long)(n0 + l15) * ldb + (hi << 3);

  v8f acc = {};
  for (int k = 0; k < K; k += 32) {
    FragBF fa, fb;
    fa.u[0] = *(const uint4*)(arow + k);
    fa.u[1] = *(const uint4*)(arow + k + 16);
    fb.u[0] = *(const uint4*)(brow + k);
    fb.u[1] = *(const uint4*)(brow + k + 16);
    acc = __builtin_amdgcn_wmma_f32_16x16x32_bf16(
        false, fa.v, false, fb.v, (short)0, acc, false, false);
  }

  const int n  = n0 + l15;
  const int mb = m0 + (hi << 3);
  const float bn = bias ? bias[n] : 0.0f;
#pragma unroll
  for (int r = 0; r < 8; ++r)
    C[(long)(mb + r) * N + n] = acc[r] + bn;
}

// fp32 [K,N] -> bf16 transposed [N, ldDst] at column offset kOff
__global__ void cvt_transpose(const float* __restrict__ W,
                              unsigned short* __restrict__ dst,
                              int K, int N_, long ldDst, long kOff)
{
  long i = (long)blockIdx.x * blockDim.x + threadIdx.x;
  if (i >= (long)K * N_) return;
  int k = (int)(i / N_);
  int n = (int)(i % N_);
  dst[(long)n * ldDst + kOff + k] = f2bf(W[i]);
}

__global__ void embed_bf16(const int* __restrict__ tokens,
                           const float* __restrict__ emb,
                           unsigned short* __restrict__ x)
{
  long i = (long)blockIdx.x * blockDim.x + threadIdx.x;
  if (i >= (long)Bc * Tc * Vc) return;
  int v = (int)(i % Vc);
  long bt = i / Vc;
  int tok = tokens[bt];
  x[i] = f2bf(emb[(long)tok * Vc + v]);
}

// latent = c3 @ w_mean + b_mean + exp((c3 @ w_sigma + b_sigma)/2) + eps
__global__ void latent_head(const float* __restrict__ c3,
                            const float* __restrict__ wm, const float* __restrict__ bm,
                            const float* __restrict__ wsg, const float* __restrict__ bs,
                            const float* __restrict__ eps,
                            unsigned short* __restrict__ latent)
{
  int idx = blockIdx.x * blockDim.x + threadIdx.x;
  if (idx >= Bc * Lc) return;
  int b = idx / Lc;
  int l = idx % Lc;
  const float* cr = c3 + (long)b * Uc;
  float m = bm[l], s = bs[l];
  for (int k = 0; k < Uc; ++k) {
    float cv = cr[k];
    m += cv * wm[(long)k * Lc + l];
    s += cv * wsg[(long)k * Lc + l];
  }
  latent[idx] = f2bf(m + expf(s * 0.5f) + eps[idx]);
}

// one wave per row; argmax over V=512, first-index tie-break (matches jnp.argmax)
__global__ void argmax_rows(const float* __restrict__ logits, int* __restrict__ out)
{
  int row = blockIdx.x * (blockDim.x >> 5) + (threadIdx.x >> 5);
  int lane = threadIdx.x & 31;
  const float* lr = logits + (long)row * Vc;
  float best = -INFINITY;
  int bi = 0x7fffffff;
  for (int v = lane; v < Vc; v += 32) {
    float x = lr[v];
    if (x > best) { best = x; bi = v; }
  }
  for (int off = 16; off; off >>= 1) {
    float ob = __shfl_xor(best, off, 32);
    int   oi = __shfl_xor(bi, off, 32);
    if (ob > best || (ob == best && oi < bi)) { best = ob; bi = oi; }
  }
  if (lane == 0) out[row] = bi;
}

__global__ void zero_u32(unsigned int* __restrict__ p, long n)
{
  long i = (long)blockIdx.x * blockDim.x + threadIdx.x;
  if (i < n) p[i] = 0u;
}

// ---------------------------------------------------------------------------
extern "C" void kernel_launch(void* const* d_in, const int* in_sizes, int n_in,
                              void* d_out, int out_size, void* d_ws, size_t ws_size,
                              hipStream_t stream)
{
  (void)in_sizes; (void)n_in; (void)out_size; (void)ws_size;

  const int*   tokens = (const int*)d_in[0];
  const float* emb    = (const float*)d_in[1];
  const float* encWx[3] = {(const float*)d_in[2],  (const float*)d_in[5],  (const float*)d_in[8]};
  const float* encWh[3] = {(const float*)d_in[3],  (const float*)d_in[6],  (const float*)d_in[9]};
  const float* encB [3] = {(const float*)d_in[4],  (const float*)d_in[7],  (const float*)d_in[10]};
  const float* decWx[3] = {(const float*)d_in[11], (const float*)d_in[14], (const float*)d_in[17]};
  const float* decWh[3] = {(const float*)d_in[12], (const float*)d_in[15], (const float*)d_in[18]};
  const float* decB [3] = {(const float*)d_in[13], (const float*)d_in[16], (const float*)d_in[19]};
  const float* w_mean  = (const float*)d_in[20];
  const float* b_mean  = (const float*)d_in[21];
  const float* w_sigma = (const float*)d_in[22];
  const float* b_sigma = (const float*)d_in[23];
  const float* dec_w   = (const float*)d_in[24];
  const float* dec_b   = (const float*)d_in[25];
  const float* eps     = (const float*)d_in[26];

  int*   outIdx    = (int*)d_out;                       // [B,T] argmax (int32)
  float* outLogits = (float*)d_out + (size_t)Bc * Tc;   // [B,T,V] logits (f32)

  // ---- workspace layout (all 256B aligned) ----
  size_t off = 0;
  auto walloc = [&](size_t bytes) -> void* {
    off = (off + 255) & ~(size_t)255;
    void* p = (char*)d_ws + off;
    off += bytes;
    return p;
  };
  // enc (1536+2048+2048)*4096 and dec (1792+2048+2048)*4096 share one region
  const long eOff[3] = {0, (long)1536 * 4096, (long)1536 * 4096 + (long)2048 * 4096};
  const int  eK  [3] = {1536, 2048, 2048};
  const long dOff[3] = {0, (long)1792 * 4096, (long)1792 * 4096 + (long)2048 * 4096};
  const int  dK  [3] = {1792, 2048, 2048};
  unsigned short* wPack = (unsigned short*)walloc((size_t)(1792 + 2048 + 2048) * 4096 * 2);
  unsigned short* wOut  = (unsigned short*)walloc((size_t)Vc * Uc * 2);          // dec_w^T
  unsigned short* xBuf  = (unsigned short*)walloc((size_t)Bc * Tc * Vc * 2);     // emb[tokens] bf16
  unsigned short* yBuf  = (unsigned short*)walloc((size_t)Bc * Tc * Uc * 2);     // decoder top h
  unsigned short* hAll  = (unsigned short*)walloc((size_t)6 * Bc * Uc * 2);      // h state x6
  float*          cAll  = (float*)walloc((size_t)6 * Bc * Uc * 4);               // c state x6
  unsigned short* latent = (unsigned short*)walloc((size_t)Bc * Lc * 2);

  unsigned short* h[6];
  float* c[6];
  for (int i = 0; i < 6; ++i) { h[i] = hAll + (size_t)i * Bc * Uc; c[i] = cAll + (size_t)i * Bc * Uc; }

  // ---- zero recurrent state (every call; harness does not re-init) ----
  {
    long nh = (long)6 * Bc * Uc / 2;  // ushorts -> u32 count
    long nc = (long)6 * Bc * Uc;
    zero_u32<<<(int)((nh + 255) / 256), 256, 0, stream>>>((unsigned int*)hAll, nh);
    zero_u32<<<(int)((nc + 255) / 256), 256, 0, stream>>>((unsigned int*)cAll, nc);
  }

  auto cvt = [&](const float* W, int K, int N_, unsigned short* dst, long ld, long kOff) {
    long tot = (long)K * N_;
    cvt_transpose<<<(int)((tot + 255) / 256), 256, 0, stream>>>(W, dst, K, N_, ld, kOff);
  };

  // ---- embedding gather -> bf16 x[B,T,V] ----
  {
    long tot = (long)Bc * Tc * Vc;
    embed_bf16<<<(int)((tot + 255) / 256), 256, 0, stream>>>(tokens, emb, xBuf);
  }

  // ---- pack encoder weights: Bt[n][k] over [Wx ; Wh], bf16, L2-resident ----
  const int encIn[3] = {Vc, Uc, Uc};
  for (int l = 0; l < 3; ++l) {
    cvt(encWx[l], encIn[l], NGATE, wPack + eOff[l], eK[l], 0);
    cvt(encWh[l], Uc,       NGATE, wPack + eOff[l], eK[l], encIn[l]);
  }

  const dim3 gStep(Bc / 16, Uc / 64);   // (8, 16): 128 blocks, 4 waves each

  // ---- encoder: 256 sequential timesteps, one fused kernel per layer ----
  for (int t = 0; t < Tc; ++t) {
    const unsigned short* xt = xBuf + (size_t)t * Vc;   // row b at b*T*V, lda=T*V
    lstm_step_wmma<<<gStep, 128, 0, stream>>>(
        xt, (long)Tc * Vc, Vc, h[0], Uc, Uc, nullptr, 0, 0,
        wPack + eOff[0], eK[0], encB[0], c[0], h[0], nullptr, 0);
    lstm_step_wmma<<<gStep, 128, 0, stream>>>(
        h[0], Uc, Uc, h[1], Uc, Uc, nullptr, 0, 0,
        wPack + eOff[1], eK[1], encB[1], c[1], h[1], nullptr, 0);
    lstm_step_wmma<<<gStep, 128, 0, stream>>>(
        h[1], Uc, Uc, h[2], Uc, Uc, nullptr, 0, 0,
        wPack + eOff[2], eK[2], encB[2], c[2], h[2], nullptr, 0);
  }

  // ---- latent head (projects the CELL state c3, per reference) ----
  latent_head<<<(Bc * Lc + 255) / 256, 256, 0, stream>>>(
      c[2], w_mean, b_mean, w_sigma, b_sigma, eps, latent);

  // ---- pack decoder weights (reuse wPack region; stream order makes it safe) ----
  const int decIn0 = Lc + Vc;
  cvt(decWx[0], decIn0, NGATE, wPack + dOff[0], dK[0], 0);
  cvt(decWh[0], Uc,     NGATE, wPack + dOff[0], dK[0], decIn0);
  for (int l = 1; l < 3; ++l) {
    cvt(decWx[l], Uc, NGATE, wPack + dOff[l], dK[l], 0);
    cvt(decWh[l], Uc, NGATE, wPack + dOff[l], dK[l], Uc);
  }
  cvt(dec_w, Uc, Vc, wOut, Uc, 0);   // output projection, transposed

  // ---- decoder: input = [latent | x_t | h3]; top layer streams h into yBuf ----
  for (int t = 0; t < Tc; ++t) {
    const unsigned short* xt = xBuf + (size_t)t * Vc;
    lstm_step_wmma<<<gStep, 128, 0, stream>>>(
        latent, Lc, Lc, xt, (long)Tc * Vc, Vc, h[3], Uc, Uc,
        wPack + dOff[0], dK[0], decB[0], c[3], h[3], nullptr, 0);
    lstm_step_wmma<<<gStep, 128, 0, stream>>>(
        h[3], Uc, Uc, h[4], Uc, Uc, nullptr, 0, 0,
        wPack + dOff[1], dK[1], decB[1], c[4], h[4], nullptr, 0);
    lstm_step_wmma<<<gStep, 128, 0, stream>>>(
        h[4], Uc, Uc, h[5], Uc, Uc, nullptr, 0, 0,
        wPack + dOff[2], dK[2], decB[2], c[5], h[5],
        yBuf + (size_t)t * Uc, (long)Tc * Uc);
  }

  // ---- logits = y @ dec_w + dec_b : one big WMMA GEMM [32768 x 512 x 1024] ----
  {
    dim3 g((Bc * Tc) / 16, Vc / 64);   // (2048, 8)
    wmma_gemm_bf16<<<g, 128, 0, stream>>>(
        yBuf, Uc, Uc, wOut, Uc, dec_b, outLogits, Vc);
  }

  // ---- argmax over V (softmax is monotonic -> argmax(logits)) ----
  argmax_rows<<<(Bc * Tc) / 8, 256, 0, stream>>>(outLogits, outIdx);
}
